// McMotLoss_47545287967224
// MI455X (gfx1250) — matrix operand order repaired and checked by reading, not compiled, not run on hardware
//
#include <hip/hip_runtime.h>
#include <hip/hip_bf16.h>
#include <math.h>

typedef __attribute__((ext_vector_type(16))) _Float16 v16h;
typedef __attribute__((ext_vector_type(8)))  float    v8f;

#define B_    4
#define C_    4
#define H_    152
#define W_    272
#define D_    128
#define NID_  200
#define NIDP_ 208          // NID padded to multiple of 16
#define K_    128
#define HW_   (H_ * W_)    // 41344
#define NPIX_ (B_ * HW_)   // 165376
#define NGROUP_ (NPIX_ / 16) // 10336, exact

// workspace float accumulators:
// 0:pos 1:neg 2:num_pos 3:wh_num 4:mask_sum 5:off_num
// 6..9:ce_sum[c] 10..13:n_valid[c] 14..17:n_sel[c]
#define WS_ACC_FLOATS 64   // 256 bytes, keeps Wh 32B-aligned

__device__ __forceinline__ float wred(float v) {
#pragma unroll
    for (int o = 16; o > 0; o >>= 1) v += __shfl_xor(v, o, 32);
    return v;
}

__global__ void k_zero(float* ws) {
    if (threadIdx.x < WS_ACC_FLOATS) ws[threadIdx.x] = 0.f;
}

// convert W_cls (C,200,128) f32 -> Wh (C,208,128) f16, zero-padded rows
__global__ void k_prepw(const float* __restrict__ W, _Float16* __restrict__ Wh) {
    int i = blockIdx.x * 256 + threadIdx.x;       // grid sized exactly
    int c = i / (NIDP_ * D_);
    int r = i % (NIDP_ * D_);
    int n = r / D_;
    int k = r % D_;
    float v = (n < NID_) ? W[(c * NID_ + n) * D_ + k] : 0.f;
    Wh[i] = (_Float16)v;
}

// focal loss partial sums (exactly one element per thread)
__global__ void k_focal(const float* __restrict__ hm, const float* __restrict__ gt,
                        float* ws) {
    int i = blockIdx.x * blockDim.x + threadIdx.x;
    float pos = 0.f, neg = 0.f, np = 0.f;
    float x = hm[i];
    float p = 1.f / (1.f + __expf(-x));
    p = fminf(fmaxf(p, 1e-4f), 1.f - 1e-4f);
    float g = gt[i];
    if (g == 1.f) {
        float om = 1.f - p;
        pos = __logf(p) * om * om;
        np = 1.f;
    } else {
        float gw = 1.f - g; gw = gw * gw; gw = gw * gw;   // (1-g)^4
        neg = __logf(1.f - p) * p * p * gw;
    }
    pos = wred(pos); neg = wred(neg); np = wred(np);
    if ((threadIdx.x & 31) == 0) {
        atomicAdd(ws + 0, pos);
        atomicAdd(ws + 1, neg);
        atomicAdd(ws + 2, np);
    }
}

// masked L1 for wh and reg: one thread per (b,k)
__global__ void k_regl1(const float* __restrict__ wh_out, const float* __restrict__ reg_out,
                        const float* __restrict__ wh_gt, const float* __restrict__ reg_gt,
                        const float* __restrict__ mask, const int* __restrict__ ind,
                        float* ws) {
    int t = threadIdx.x;                 // 0..511
    int b = t / K_, k = t % K_;
    float m = mask[b * K_ + k];
    int id = ind[b * K_ + k];
    float wnum = 0.f, onum = 0.f;
#pragma unroll
    for (int ch = 0; ch < 2; ++ch) {
        float pw = wh_out[(b * 2 + ch) * HW_ + id];
        float po = reg_out[(b * 2 + ch) * HW_ + id];
        wnum += fabsf(pw * m - wh_gt[(b * K_ + k) * 2 + ch] * m);
        onum += fabsf(po * m - reg_gt[(b * K_ + k) * 2 + ch] * m);
    }
    wnum = wred(wnum); onum = wred(onum); float ms = wred(m);
    if ((t & 31) == 0) {
        atomicAdd(ws + 3, wnum);
        atomicAdd(ws + 5, onum);
        atomicAdd(ws + 4, ms);
    }
}

// ReID: one wave32 per 16-pixel group; 4 waves per block.
// Per group: normalize 16 embeddings into f16 A-fragments, then for each of
// 4 classes compute a 16x208 logits tile with v_wmma_f32_16x16x32_f16
// (13 N-tiles x 4 K-steps), logsumexp + masked CE from LDS.
__global__ void __launch_bounds__(128)
k_reid(const float* __restrict__ id_out, const _Float16* __restrict__ Wh,
       const float* __restrict__ b_cls, const int* __restrict__ clsmap,
       const int* __restrict__ trids, float* ws) {
    __shared__ float lds[4][16][NIDP_];

    const int lane  = threadIdx.x & 31;
    const int wv    = threadIdx.x >> 5;
    const int group = blockIdx.x * 4 + wv;       // grid divides exactly
    const int row   = lane & 15;
    const int hi    = lane >> 4;
    const int p     = group * 16 + row;
    const int b     = p / HW_;
    const int hw    = p % HW_;
    const float kScale = 1.4142135623730951f * logf((float)(NID_ - 1));

    // pass 1: sum of squares of this pixel's 128-d embedding
    // (lane pair (row,row+16) holds complementary halves -> combine via shfl)
    float sq = 0.f;
#pragma unroll
    for (int s = 0; s < 4; ++s) {
#pragma unroll
        for (int j = 0; j < 16; ++j) {
            int kk = 32 * s + 16 * hi + j; // contiguous K order, same as B-frag
            float v = id_out[(b * D_ + kk) * HW_ + hw];
            sq += v * v;
        }
    }
    sq += __shfl_xor(sq, 16, 32);
    float inv = kScale / fmaxf(sqrtf(sq), 1e-12f);

    // pass 2: build scaled f16 A-fragments (cache-hot reload)
    v16h a[4];
#pragma unroll
    for (int s = 0; s < 4; ++s) {
#pragma unroll
        for (int j = 0; j < 16; ++j) {
            int kk = 32 * s + 16 * hi + j;
            float v = id_out[(b * D_ + kk) * HW_ + hw];
            a[s][j] = (_Float16)(v * inv);
        }
    }

    const int myc = clsmap[b * HW_ + hw];   // selected class for this pixel

    for (int c = 0; c < C_; ++c) {
        // 13 N-tiles of 16 columns each
        for (int t = 0; t < 13; ++t) {
            v8f acc = {};
#pragma unroll
            for (int s = 0; s < 4; ++s) {
                // B-frag: lane holds 16 contiguous K for column n = lane&15
                const _Float16* bp =
                    Wh + (((size_t)c * NIDP_ + t * 16 + (lane & 15)) * D_ + 32 * s + 16 * hi);
                v16h bf = *(const v16h*)bp;
                acc = __builtin_amdgcn_wmma_f32_16x16x32_f16(
                    false, a[s], false, bf, (short)0, acc, false, false);
            }
            int ncol  = t * 16 + (lane & 15);
            float bias = (ncol < NID_) ? b_cls[c * NID_ + ncol] : -1e30f;
#pragma unroll
            for (int i = 0; i < 8; ++i)        // D layout: row = 8*hi + i
                lds[wv][hi * 8 + i][ncol] = acc[i] + bias;
        }
        __syncthreads();

        // logsumexp over 208 cols: lane pair splits columns 104/104
        const int c0 = hi * 104;
        float m = -3e38f;
        for (int n = 0; n < 104; ++n) m = fmaxf(m, lds[wv][row][c0 + n]);
        float sm = 0.f;
        for (int n = 0; n < 104; ++n) sm += __expf(lds[wv][row][c0 + n] - m);
        float m2 = __shfl_xor(m, 16, 32);
        float s2 = __shfl_xor(sm, 16, 32);
        float M  = fmaxf(m, m2);
        float S  = sm * __expf(m - M) + s2 * __expf(m2 - M);
        float lse = M + __logf(S);

        float ce = 0.f, nval = 0.f, nsel = 0.f;
        if (hi == 0) {
            int tgt   = trids[((b * C_) + c) * HW_ + hw];
            bool sel   = (myc == c);
            bool valid = sel && (tgt >= 0);
            nsel = sel ? 1.f : 0.f;
            nval = valid ? 1.f : 0.f;
            ce   = valid ? (lse - lds[wv][row][tgt]) : 0.f;
        }
        ce = wred(ce); nval = wred(nval); nsel = wred(nsel);
        if (lane == 0) {
            atomicAdd(ws + 6 + c, ce);
            atomicAdd(ws + 10 + c, nval);
            atomicAdd(ws + 14 + c, nsel);
        }
        __syncthreads();   // protect LDS before next class overwrites it
    }
}

__global__ void k_final(const float* __restrict__ ws, const float* __restrict__ s_det,
                        const float* __restrict__ s_id, float* __restrict__ out) {
    float pos = ws[0], neg = ws[1], np = ws[2];
    float hm = (np > 0.f) ? (-(pos + neg) / fmaxf(np, 1.f)) : (-neg);
    float den = ws[4] * 2.f + 1e-4f;
    float wh  = ws[3] / den;
    float off = ws[5] / den;
    float reid = 0.f;
    for (int c = 0; c < C_; ++c) {
        float ns  = ws[14 + c];
        float cem = ws[6 + c] / fmaxf(ws[10 + c], 1.f);
        reid += (ns > 0.f) ? cem / fmaxf(ns, 1.f) : 0.f;
    }
    float det = 1.0f * hm + 0.1f * wh + 1.0f * off;
    float sd = s_det[0], si = s_id[0];
    float loss = 0.5f * (__expf(-sd) * det + __expf(-si) * reid + (sd + si));
    out[0] = loss; out[1] = hm; out[2] = wh; out[3] = off; out[4] = reid;
}

extern "C" void kernel_launch(void* const* d_in, const int* in_sizes, int n_in,
                              void* d_out, int out_size, void* d_ws, size_t ws_size,
                              hipStream_t stream) {
    const float* hm_out   = (const float*)d_in[0];
    const float* wh_out   = (const float*)d_in[1];
    const float* reg_out  = (const float*)d_in[2];
    const float* id_out   = (const float*)d_in[3];
    const float* hm_gt    = (const float*)d_in[4];
    const float* wh_gt    = (const float*)d_in[5];
    const float* reg_gt   = (const float*)d_in[6];
    const float* reg_mask = (const float*)d_in[7];
    const float* W_cls    = (const float*)d_in[8];
    const float* b_cls    = (const float*)d_in[9];
    const float* s_det    = (const float*)d_in[10];
    const float* s_id     = (const float*)d_in[11];
    const int*   ind      = (const int*)d_in[12];
    const int*   clsmap   = (const int*)d_in[13];
    const int*   trids    = (const int*)d_in[14];

    float*     ws = (float*)d_ws;
    _Float16*  Wh = (_Float16*)((char*)d_ws + WS_ACC_FLOATS * sizeof(float));

    k_zero <<<1, 64, 0, stream>>>(ws);
    k_prepw<<<(C_ * NIDP_ * D_) / 256, 256, 0, stream>>>(W_cls, Wh);
    k_focal<<<(B_ * C_ * HW_) / 256, 256, 0, stream>>>(hm_out, hm_gt, ws);
    k_regl1<<<1, B_ * K_, 0, stream>>>(wh_out, reg_out, wh_gt, reg_gt, reg_mask, ind, ws);
    k_reid <<<NGROUP_ / 4, 128, 0, stream>>>(id_out, Wh, b_cls, clsmap, trids, ws);
    k_final<<<1, 1, 0, stream>>>(ws, s_det, s_id, (float*)d_out);
}
